// SparseStudentSNN1d_67834713473599
// MI455X (gfx1250) — compile-verified, matrix operand order checked by reading, not run on hardware
//
#include <hip/hip_runtime.h>

// ---------------------------------------------------------------------------
// SparseStudentSNN1d on gfx1250.
// Both convs run as per-tap f16 WMMA GEMMs with f32 accumulation.
// LDS panels are column-major (channel contiguous) so every WMMA fragment is
// two contiguous 16B chunks -> ds_load_b128; spike writeback is ds_store_b128.
// v2 membrane / spike-count recurrence state lives in registers across T.
// w2 fragments live in LDS (reloaded as b128 pairs); w1 fragments + biases are
// wave-invariant and preloaded once. All hot-loop indexing is shift/mask.
// ---------------------------------------------------------------------------

#define TILE_L  64
#define NTILES  4            // TILE_L / 16
#define S1COLS  80           // 5 n-tiles: s1 cols [l0-8, l0+72)
#define S1STR   72           // halves per s1 column (144B, 16B-aligned)
#define XCOLS   82           // x cols [l0-9, l0+73)
#define XSTR2   40           // halves per x column (80B, 16B-aligned)
#define W1STR   40           // halves per w1 row
#define W2STR   72           // halves per w2 row (row = c2, 64 k + pad)
#define CIN     12
#define C1      64
#define C2      128
#define LLEN    2048
#define TSTEPS  20
#define NCLS    4
#define TAU2INV (1.0f / 0.9f)

typedef _Float16 h16;
typedef __attribute__((ext_vector_type(16))) _Float16 v16h;
typedef __attribute__((ext_vector_type(8)))  _Float16 v8h;
typedef __attribute__((ext_vector_type(8)))  float    v8f;

union F16Frag { v16h v; v8h h[2]; };

__global__ void snn_zero_kernel(float* __restrict__ p, int n) {
    int i = blockIdx.x * 256 + threadIdx.x;
    if (i < n) p[i] = 0.0f;
}

__global__ __launch_bounds__(256)
void snn_main_kernel(const float* __restrict__ x,
                     const float* __restrict__ w1,
                     const float* __restrict__ b1,
                     const float* __restrict__ th1p,
                     const float* __restrict__ w2,
                     const float* __restrict__ b2,
                     const float* __restrict__ th2p,
                     const float* __restrict__ gainp,
                     float* __restrict__ pooled)
{
    // Column-major panels: [col][channel], channel contiguous.
    __shared__ h16   xp [XCOLS  * XSTR2];   // ~6.6 KB  x tile, f16
    __shared__ h16   s1p[S1COLS * S1STR];   // ~11.5 KB binary spikes, f16
    __shared__ h16   w1p01[C1 * W1STR];     // taps 0(+k<16) / 1(k>=16)
    __shared__ h16   w1p2 [C1 * W1STR];     // tap 2 (upper half zero)
    __shared__ h16   w2p[3 * C2 * W2STR];   // ~55 KB  w2, per-tap, row-major
    __shared__ float pool[C2];

    const int tid  = threadIdx.x;
    const int wave = tid >> 5;
    const int lane = tid & 31;
    const int lh   = lane >> 4;             // lane half
    const int ln   = lane & 15;
    const int hofs = lh << 3;               // 8*lh
    const int b    = blockIdx.y;
    const int l0   = blockIdx.x * TILE_L;
    const int ls   = l0 - 8;                // l of s1p column 0
    const int xbase = l0 - 9;               // l of xp column 0

    // ---- stage w1 panels (tap-packed, zero-padded), zero xp --------------
#pragma unroll 1
    for (int i = tid; i < C1 * W1STR; i += 256) {
        int m = i / W1STR, k = i % W1STR;
        float v01 = 0.0f, v2w = 0.0f;
        if (k < CIN) { v01 = w1[(m * CIN + k) * 3 + 0];
                       v2w = w1[(m * CIN + k) * 3 + 2]; }
        else if (k >= 16 && k < 16 + CIN)
                       v01 = w1[(m * CIN + (k - 16)) * 3 + 1];
        w1p01[i] = (h16)v01;
        w1p2[i]  = (h16)v2w;
    }
#pragma unroll 1
    for (int i = tid; i < XCOLS * XSTR2; i += 256) xp[i] = (h16)0.0f;

    // ---- stage w2 per-tap panels: w2p[dk][c2][c1] (shift/mask only) -------
#pragma unroll 1
    for (int i = tid; i < 3 * C2 * C1; i += 256) {
        int dk  = i >> 13;                  // / (128*64)
        int rem = i & 8191;                 // c2*64 + c1
        int m   = rem >> 6, k = rem & 63;
        w2p[dk * (C2 * W2STR) + m * W2STR + k] = (h16)w2[rem * 3 + dk];
    }

    const float gain = gainp[0];
    const float th1  = th1p[0];
    const float th2  = th2p[0];
    const float Ac   = 1.0f - TAU2INV;      // v2 decay
    const float Bc   = gain * TAU2INV;      // h2raw coefficient

    // per-wave constants (GEMM1 m-tile is wave & 3; GEMM2 c2 stripe is wave)
    const int mt1  = wave & 3;
    const int ntlo = wave >> 2;
    const int m1   = mt1 * 16 + ln;         // A fragment: M = lane % 16
    const int mrow = wave * 16 + ln;        // w2 row for this lane

    float b1f[8], Cf[8];
#pragma unroll
    for (int r = 0; r < 8; ++r) {
        b1f[r] = b1[mt1 * 16 + hofs + r];
        Cf[r]  = b2[wave * 16 + hofs + r] * gain * TAU2INV;
    }

    // ---- persistent recurrence state --------------------------------------
    v8f v2f[NTILES], accf[NTILES];
    v8f vzero = {};
#pragma unroll
    for (int nt = 0; nt < NTILES; ++nt) { v2f[nt] = vzero; accf[nt] = vzero; }

    // x-staging lane decomposition (no div/mod in the hot loop)
    const int sci  = tid & 15;              // channel 0..15 (12 active)
    const int scol = tid >> 4;              // column group base

    __syncthreads();

    // wave-invariant GEMM1 A-fragments, preloaded once (stay in registers)
    F16Frag a01, a2f;
    a01.h[0] = *(const v8h*)&w1p01[m1 * W1STR + hofs];
    a01.h[1] = *(const v8h*)&w1p01[m1 * W1STR + 16 + hofs];
    a2f.h[0] = *(const v8h*)&w1p2 [m1 * W1STR + hofs];
    a2f.h[1] = *(const v8h*)&w1p2 [m1 * W1STR + 16 + hofs];

    // one conv1 output tile: GEMM (2 wmma) + threshold + b128 spike store
    auto conv1_tile = [&](int nt1) {
        int n = nt1 * 16 + ln;              // B fragment: N = lane % 16
        F16Frag b01, b2v;
        // taps 0/1 packed in one K=32 fragment; tap 2 upper half is zero
        b01.h[0] = *(const v8h*)&xp[(n + 0) * XSTR2 + hofs];
        b01.h[1] = *(const v8h*)&xp[(n + 1) * XSTR2 + hofs];
        b2v.h[0] = *(const v8h*)&xp[(n + 2) * XSTR2 + hofs];
        b2v.h[1] = *(const v8h*)&xp[(n + 2) * XSTR2 + 16 + hofs];   // zeros
        v8f c = vzero;
        c = __builtin_amdgcn_wmma_f32_16x16x32_f16(
                false, a01.v, false, b01.v, (short)0, c, false, false);
        c = __builtin_amdgcn_wmma_f32_16x16x32_f16(
                false, a2f.v, false, b2v.v, (short)0, c, false, false);
        int l = ls + n;
        bool inr = (l >= 0) && (l < LLEN);
        v8h sv;
#pragma unroll
        for (int r = 0; r < 8; ++r) {
            float hv = (c[r] + b1f[r]) * gain;
            sv[r] = (inr && (hv >= th1)) ? (h16)1.0f : (h16)0.0f;
        }
        *(v8h*)&s1p[n * S1STR + mt1 * 16 + hofs] = sv;
    };

    // ======================= timestep loop (sequential) ====================
    for (int t = 0; t < TSTEPS; ++t) {
        // stage x[:, :, :, t] tile into column-major f16 panel
        if (sci < CIN) {
#pragma unroll
            for (int p = 0; p < 6; ++p) {           // 6*16 = 96 >= 82 cols
                int col = scol + p * 16;
                if (col < XCOLS) {
                    int l = xbase + col;
                    float v = 0.0f;
                    if (l >= 0 && l < LLEN)
                        v = x[(((size_t)b * CIN + sci) * LLEN + l) * TSTEPS + t];
                    xp[col * XSTR2 + sci] = (h16)v;
                }
            }
        }
        __syncthreads();

        // ---- GEMM1 + threshold: division-free static tiling ---------------
        conv1_tile(ntlo);
        conv1_tile(ntlo + 2);
        if (wave < 4) conv1_tile(4);
        __syncthreads();

        // ---- GEMM2: A frags from LDS per (tap, k-tile), 4 n-tiles ---------
        v8f h2[NTILES];
#pragma unroll
        for (int nt = 0; nt < NTILES; ++nt) h2[nt] = vzero;

#pragma unroll
        for (int dk = 0; dk < 3; ++dk)
#pragma unroll
            for (int kt = 0; kt < 2; ++kt) {
                const h16* wrow = &w2p[dk * (C2 * W2STR) + mrow * W2STR + kt * 32];
                F16Frag a;
                a.h[0] = *(const v8h*)&wrow[hofs];
                a.h[1] = *(const v8h*)&wrow[16 + hofs];
#pragma unroll
                for (int nt = 0; nt < NTILES; ++nt) {
                    const h16* scol2 =
                        &s1p[(nt * 16 + ln + dk + 7) * S1STR + kt * 32];
                    F16Frag bb;
                    bb.h[0] = *(const v8h*)&scol2[hofs];
                    bb.h[1] = *(const v8h*)&scol2[16 + hofs];
                    h2[nt] = __builtin_amdgcn_wmma_f32_16x16x32_f16(
                                 false, a.v, false, bb.v, (short)0,
                                 h2[nt], false, false);
                }
            }

        // ---- leaky integrate (2 FMA), spike, reset, accumulate ------------
#pragma unroll
        for (int nt = 0; nt < NTILES; ++nt)
#pragma unroll
            for (int r = 0; r < 8; ++r) {
                float v2 = fmaf(v2f[nt][r], Ac, fmaf(h2[nt][r], Bc, Cf[r]));
                if (v2 >= th2) { accf[nt][r] += 1.0f; v2 = 0.0f; }
                v2f[nt][r] = v2;
            }
        __syncthreads();
    }

    // ---- reduce spike counts over L in-block, then to global --------------
    if (tid < C2) pool[tid] = 0.0f;
    __syncthreads();
#pragma unroll
    for (int nt = 0; nt < NTILES; ++nt)
#pragma unroll
        for (int r = 0; r < 8; ++r)
            atomicAdd(&pool[wave * 16 + hofs + r], accf[nt][r]);
    __syncthreads();
    if (tid < C2) atomicAdd(&pooled[b * C2 + tid], pool[tid]);
}

__global__ void snn_fc_kernel(const float* __restrict__ pooled,
                              const float* __restrict__ fc_w,
                              const float* __restrict__ fc_b,
                              float* __restrict__ out)
{
    int i = threadIdx.x;                    // 128 threads: (b, class) pairs
    if (i < 32 * NCLS) {
        int bb = i / NCLS, nc = i % NCLS;
        float s = 0.0f;
        for (int c = 0; c < C2; ++c)
            s += pooled[bb * C2 + c] * fc_w[nc * C2 + c];
        out[i] = s * (1.0f / ((float)TSTEPS * (float)LLEN)) + fc_b[nc];
    }
}

extern "C" void kernel_launch(void* const* d_in, const int* in_sizes, int n_in,
                              void* d_out, int out_size, void* d_ws, size_t ws_size,
                              hipStream_t stream) {
    const float* x    = (const float*)d_in[0];
    const float* w1   = (const float*)d_in[1];
    const float* b1   = (const float*)d_in[2];
    const float* th1  = (const float*)d_in[3];
    const float* w2   = (const float*)d_in[4];
    const float* b2   = (const float*)d_in[5];
    const float* th2  = (const float*)d_in[6];
    const float* gain = (const float*)d_in[7];
    const float* fcw  = (const float*)d_in[8];
    const float* fcb  = (const float*)d_in[9];

    float* pooled = (float*)d_ws;           // 32*128 floats of scratch
    float* out    = (float*)d_out;          // 32*4 logits

    snn_zero_kernel<<<dim3((32 * C2 + 255) / 256), 256, 0, stream>>>(pooled, 32 * C2);
    snn_main_kernel<<<dim3(LLEN / TILE_L, 32), 256, 0, stream>>>(
        x, w1, b1, th1, w2, b2, th2, gain, pooled);
    snn_fc_kernel<<<1, 128, 0, stream>>>(pooled, fcw, fcb, out);
}
